// GNN_encoder_38491496907194
// MI455X (gfx1250) — compile-verified, matrix operand order checked by reading, not run on hardware
//
#include <hip/hip_runtime.h>
#include <hip/hip_bf16.h>
#include <math.h>

// ---------------- problem constants ----------------
#define NN   50000     // nodes
#define NE   100000    // edges
#define FIN  32
#define FE   8
#define F1O  64
#define F2O  64
#define KHH  32
#define FCO  32
#define KEXT 1056      // 32*32 (k,i) pairs + 32 bias rows

typedef __attribute__((ext_vector_type(16))) __bf16 v16bf;
typedef __attribute__((ext_vector_type(8)))  __bf16 v8bf;
typedef __attribute__((ext_vector_type(8)))  float  v8f;

// ---------------- workspace layout (bytes) ----------------
// Bt   : bf16 [64][1056]  (B^T for edge GEMM, incl bias rows)   135168
// rootT: bf16 [64][32]                                            4096
// gatT : bf16 [64][64]                                            8192
// agg  : f32 [N][64]      ECC segment-sum accumulator
// x1   : f32 [N][64]
// xp   : f32 [N][64]
// outA : f32 [N][64]      GAT output accumulator
// as/an: f32 [N]          attention logits
// smK  : u32 [N]          segment-max keys (monotonic float encoding)
// den  : f32 [N]
// sc   : f32 [E]          edge scores
// ex   : f32 [E]
// pool : f32 [64]
static const size_t O_BT   = 0;
static const size_t O_ROOT = 135168;
static const size_t O_GAT  = O_ROOT + 4096;
static const size_t O_AGG  = 147456;
static const size_t O_X1   = O_AGG + (size_t)NN*64*4;
static const size_t O_XP   = O_X1  + (size_t)NN*64*4;
static const size_t O_OUT  = O_XP  + (size_t)NN*64*4;
static const size_t O_AS   = O_OUT + (size_t)NN*64*4;
static const size_t O_AN   = O_AS  + (size_t)NN*4;
static const size_t O_SMK  = O_AN  + (size_t)NN*4;
static const size_t O_DEN  = O_SMK + (size_t)NN*4;
static const size_t O_SC   = O_DEN + (size_t)NN*4;
static const size_t O_EX   = O_SC  + (size_t)NE*4;
static const size_t O_POOL = O_EX  + (size_t)NE*4;

// ---------------- helpers ----------------
__device__ __forceinline__ v8f wmma_bf16(v16bf a, v16bf b, v8f c) {
    return __builtin_amdgcn_wmma_f32_16x16x32_bf16(
        /*neg_a=*/false, a, /*neg_b=*/false, b,
        /*c_mod=*/(short)0, c, /*reuse_a=*/false, /*reuse_b=*/false);
}

// Load a 32x16 B fragment (16-bit layout) from a [rows][rowlen] bf16 matrix
// stored K-contiguous per output column o.  Lane layout: col = lane&15,
// K = koff + (e&7) + 16*(e>>3) with koff already including base=(lane>>4)*8.
__device__ __forceinline__ v16bf ldb(const __bf16* B, int rowlen, int o, int koff) {
    const v8bf lo = *(const v8bf*)(B + (size_t)o * rowlen + koff);
    const v8bf hi = *(const v8bf*)(B + (size_t)o * rowlen + koff + 16);
    v16bf r;
#pragma unroll
    for (int t = 0; t < 8; ++t) { r[t] = lo[t]; r[t + 8] = hi[t]; }
    return r;
}

// monotonic float<->uint key for atomicMax-based segment max
__device__ __forceinline__ unsigned fkey(float x) {
    unsigned u = __float_as_uint(x);
    return (u & 0x80000000u) ? ~u : (u | 0x80000000u);
}
__device__ __forceinline__ float fdec(unsigned k) {
    unsigned u = (k & 0x80000000u) ? (k & 0x7fffffffu) : ~k;
    return __uint_as_float(u);
}

// ---------------- kernel 0: weight re-layout to bf16 ----------------
// Bt[o][k*32+i]   = ecc_w1[k][i*64+o];  Bt[o][1024+i] = ecc_b1[i*64+o]
// rootT[o][i]     = ecc_root[i][o];     gatT[o][i]    = gat_kernel[i][o]
__global__ void k_prep(const float* w1, const float* b1, const float* root,
                       const float* gk, __bf16* Bt, __bf16* rootT, __bf16* gatT) {
    int gid = blockIdx.x * blockDim.x + threadIdx.x;
    if (gid < 64 * KEXT) {
        int o = gid / KEXT, ke = gid % KEXT;
        float v;
        if (ke < 1024) { int k = ke >> 5, i = ke & 31; v = w1[(size_t)k * 2048 + i * 64 + o]; }
        else           { int i = ke - 1024;            v = b1[i * 64 + o]; }
        Bt[(size_t)o * KEXT + ke] = (__bf16)v;
    }
    if (gid < 64 * 32) { int o = gid >> 5, i = gid & 31; rootT[o * 32 + i] = (__bf16)root[i * 64 + o]; }
    if (gid < 64 * 64) { int o = gid >> 6, i = gid & 63; gatT[o * 64 + i]  = (__bf16)gk[i * 64 + o]; }
}

// ---------------- kernel 1: init accumulators ----------------
__global__ void k_init(float* agg, float* outA, float* den, unsigned* smK, float* pool) {
    int gid = blockIdx.x * blockDim.x + threadIdx.x;
    if (gid < NN * 64) { agg[gid] = 0.f; outA[gid] = 0.f; }
    if (gid < NN)      { den[gid] = 0.f; smK[gid] = fkey(-3.0e38f); }
    if (gid < 64)      { pool[gid] = 0.f; }
}

// ---------------- kernel 2: ECC edge GEMM (WMMA) + scatter ----------------
// block = 256 threads (8 waves), each wave owns a 16-edge tile.
// LDS: Bt copy (135168 B) + h[8][16][33] f32 (16896 B) + dst[8][16] (512 B)
#define ECC_SMEM (135168 + 16896 + 512)
__global__ void k_ecc(const float* x, const float* e, const int* eidx,
                      const float* w0, const float* b0, const __bf16* Btg,
                      float* agg) {
    extern __shared__ char smem[];
    __bf16* sB = (__bf16*)smem;
    float*  sH = (float*)(smem + 135168);
    int*    sD = (int*)(smem + 135168 + 16896);

    // cooperative load of B (1056x64 bf16, transposed) into LDS
    const unsigned* gB = (const unsigned*)Btg;
    unsigned* sBw = (unsigned*)smem;
    for (int i = threadIdx.x; i < (64 * KEXT) / 2; i += 256) sBw[i] = gB[i];

    const int wv = threadIdx.x >> 5, lane = threadIdx.x & 31;
    const int m = lane & 15, half = lane >> 4, base = half * 8;
    const int eid = (blockIdx.x * 8 + wv) * 16 + m;
    const bool valid = (eid < NE);

    int src = valid ? eidx[eid] : 0;
    int dst = valid ? eidx[NE + eid] : 0;
    if (half == 0) sD[wv * 16 + m] = dst;

    // kernel-network hidden layer: h[m][kk] = relu(e_row @ w0 + b0)
    float er[FE];
#pragma unroll
    for (int j = 0; j < FE; ++j) er[j] = valid ? e[(size_t)eid * FE + j] : 0.f;
#pragma unroll
    for (int t = 0; t < 16; ++t) {
        int kk = half * 16 + t;
        float acc = b0[kk];
#pragma unroll
        for (int j = 0; j < FE; ++j) acc += er[j] * w0[j * 32 + kk];
        sH[wv * 528 + m * 33 + kk] = fmaxf(acc, 0.f);
    }

    // gather x[src] row into A-fragment order (f32 regs)
    float xr[16];
#pragma unroll
    for (int t = 0; t < 16; ++t) {
        int k = base + (t & 7) + ((t >> 3) << 4);
        xr[t] = valid ? x[(size_t)src * FIN + k] : 0.f;
    }

    __syncthreads();

    v8f a0 = {}, a1 = {}, a2 = {}, a3 = {};
    const __bf16* sBp = sB;
    for (int kb = 0; kb < 32; ++kb) {            // rank-1 update trick:
        float s = sH[wv * 528 + m * 33 + kb];    // A_kb = h[m,kb] * x_src row
        v16bf a;
#pragma unroll
        for (int t = 0; t < 16; ++t) a[t] = (__bf16)(xr[t] * s);
        int koff = kb * 32 + base;
        a0 = wmma_bf16(a, ldb(sBp, KEXT,  0 + m, koff), a0);
        a1 = wmma_bf16(a, ldb(sBp, KEXT, 16 + m, koff), a1);
        a2 = wmma_bf16(a, ldb(sBp, KEXT, 32 + m, koff), a2);
        a3 = wmma_bf16(a, ldb(sBp, KEXT, 48 + m, koff), a3);
    }
    {   // bias rows: + x_src @ b1_reshaped
        v16bf a;
#pragma unroll
        for (int t = 0; t < 16; ++t) a[t] = (__bf16)xr[t];
        int koff = 1024 + base;
        a0 = wmma_bf16(a, ldb(sBp, KEXT,  0 + m, koff), a0);
        a1 = wmma_bf16(a, ldb(sBp, KEXT, 16 + m, koff), a1);
        a2 = wmma_bf16(a, ldb(sBp, KEXT, 32 + m, koff), a2);
        a3 = wmma_bf16(a, ldb(sBp, KEXT, 48 + m, koff), a3);
    }

    // scatter msgs tile into agg[dst] (segment_sum)
    v8f acc[4] = {a0, a1, a2, a3};
#pragma unroll
    for (int nt = 0; nt < 4; ++nt)
#pragma unroll
        for (int j = 0; j < 8; ++j) {
            int M = j + 8 * half;
            int d = sD[wv * 16 + M];
            atomicAdd(&agg[(size_t)d * 64 + nt * 16 + m], acc[nt][j]);
        }
}

// ---------------- kernel 3: x1 = relu(agg + x@root + bias)  (WMMA) ----------
__global__ void k_node1(const float* x, const float* agg, const __bf16* rootT,
                        const float* ebias, float* x1) {
    int wv = threadIdx.x >> 5, lane = threadIdx.x & 31;
    int tile = blockIdx.x * 8 + wv;
    if (tile >= NN / 16) return;
    int m = lane & 15, half = lane >> 4, base = half * 8;
    int row = tile * 16 + m;
    v16bf a;
#pragma unroll
    for (int t = 0; t < 16; ++t) {
        int k = base + (t & 7) + ((t >> 3) << 4);
        a[t] = (__bf16)x[(size_t)row * FIN + k];
    }
    v8f acc[4];
#pragma unroll
    for (int nt = 0; nt < 4; ++nt) {
        v8f c = {};
        acc[nt] = wmma_bf16(a, ldb(rootT, 32, nt * 16 + m, base), c);
    }
#pragma unroll
    for (int nt = 0; nt < 4; ++nt)
#pragma unroll
        for (int j = 0; j < 8; ++j) {
            int grow = tile * 16 + j + 8 * half, col = nt * 16 + m;
            float v = acc[nt][j] + agg[(size_t)grow * 64 + col] + ebias[col];
            x1[(size_t)grow * 64 + col] = fmaxf(v, 0.f);
        }
}

// ---------------- kernel 4: xp = x1 @ gat_kernel  (WMMA) --------------------
__global__ void k_node2(const float* x1, const __bf16* gatT, float* xp) {
    int wv = threadIdx.x >> 5, lane = threadIdx.x & 31;
    int tile = blockIdx.x * 8 + wv;
    if (tile >= NN / 16) return;
    int m = lane & 15, half = lane >> 4, base = half * 8;
    int row = tile * 16 + m;
    v8f acc[4] = {{}, {}, {}, {}};
#pragma unroll
    for (int kb = 0; kb < 2; ++kb) {
        v16bf a;
#pragma unroll
        for (int t = 0; t < 16; ++t) {
            int k = kb * 32 + base + (t & 7) + ((t >> 3) << 4);
            a[t] = (__bf16)x1[(size_t)row * 64 + k];
        }
#pragma unroll
        for (int nt = 0; nt < 4; ++nt)
            acc[nt] = wmma_bf16(a, ldb(gatT, 64, nt * 16 + m, kb * 32 + base), acc[nt]);
    }
#pragma unroll
    for (int nt = 0; nt < 4; ++nt)
#pragma unroll
        for (int j = 0; j < 8; ++j) {
            int grow = tile * 16 + j + 8 * half, col = nt * 16 + m;
            xp[(size_t)grow * 64 + col] = acc[nt][j];
        }
}

// ---------------- kernel 5: attention logits per node -----------------------
__global__ void k_attn(const float* xp, const float* wself, const float* wneigh,
                       float* as_, float* an_) {
    int n = blockIdx.x * blockDim.x + threadIdx.x;
    if (n >= NN) return;
    float s = 0.f, t = 0.f;
#pragma unroll 8
    for (int c = 0; c < 64; ++c) {
        float v = xp[(size_t)n * 64 + c];
        s += v * wself[c]; t += v * wneigh[c];
    }
    as_[n] = s; an_[n] = t;
}

// ---------------- kernel 6: edge scores + segment max -----------------------
__global__ void k_emax(const int* eidx, const float* as_, const float* an_,
                       float* sc, unsigned* smK) {
    int ei = blockIdx.x * blockDim.x + threadIdx.x;
    if (ei >= NE) return;
    int src = eidx[ei], dst = eidx[NE + ei];
    float s = as_[dst] + an_[src];
    s = (s >= 0.f) ? s : 0.2f * s;               // leaky_relu(0.2)
    sc[ei] = s;
    atomicMax(&smK[dst], fkey(s));
}

// ---------------- kernel 7: exp + denominator -------------------------------
__global__ void k_eexp(const int* eidx, const float* sc, const unsigned* smK,
                       float* ex, float* den) {
    int ei = blockIdx.x * blockDim.x + threadIdx.x;
    if (ei >= NE) return;
    int dst = eidx[NE + ei];
    float v = __expf(sc[ei] - fdec(smK[dst]));
    ex[ei] = v;
    atomicAdd(&den[dst], v);
}

// ---------------- kernel 8: weighted scatter out += alpha * xp[src] ---------
__global__ void k_escat(const int* eidx, const float* ex, const float* den,
                        const float* xp, float* outA) {
    long long gid = (long long)blockIdx.x * blockDim.x + threadIdx.x;
    if (gid >= (long long)NE * 64) return;
    int ei = (int)(gid >> 6), c = (int)(gid & 63);
    int src = eidx[ei], dst = eidx[NE + ei];
    float alpha = ex[ei] / (den[dst] + 1e-9f);
    atomicAdd(&outA[(size_t)dst * 64 + c], alpha * xp[(size_t)src * 64 + c]);
}

// ---------------- kernel 9: relu(out+bias), pooled mean partials ------------
__global__ void k_pool(const float* outA, const float* gbias, float* pool) {
    int c = threadIdx.x;                 // 64 threads
    int n0 = blockIdx.x * 256;
    float s = 0.f;
    for (int r = 0; r < 256; ++r) {
        int n = n0 + r;
        if (n < NN) s += fmaxf(outA[(size_t)n * 64 + c] + gbias[c], 0.f);
    }
    atomicAdd(&pool[c], s);
}

// ---------------- kernel 10: final dense ------------------------------------
__global__ void k_dense(const float* pool, const float* fcw, const float* fcb,
                        float* out) {
    int f = threadIdx.x;                 // 32 threads
    if (f >= FCO) return;
    float acc = fcb[f];
#pragma unroll 8
    for (int c = 0; c < 64; ++c) acc += (pool[c] * (1.0f / NN)) * fcw[c * FCO + f];
    out[f] = fmaxf(acc, 0.f);
}

// ---------------- launcher ---------------------------------------------------
extern "C" void kernel_launch(void* const* d_in, const int* in_sizes, int n_in,
                              void* d_out, int out_size, void* d_ws, size_t ws_size,
                              hipStream_t stream) {
    (void)in_sizes; (void)n_in; (void)out_size; (void)ws_size;
    const float* x      = (const float*)d_in[0];
    const float* e      = (const float*)d_in[1];
    const int*   eidx   = (const int*)  d_in[2];
    const float* w0     = (const float*)d_in[3];
    const float* b0     = (const float*)d_in[4];
    const float* w1     = (const float*)d_in[5];
    const float* b1     = (const float*)d_in[6];
    const float* root   = (const float*)d_in[7];
    const float* ebias  = (const float*)d_in[8];
    const float* gk     = (const float*)d_in[9];
    const float* wself  = (const float*)d_in[10];
    const float* wneigh = (const float*)d_in[11];
    const float* gbias  = (const float*)d_in[12];
    const float* fcw    = (const float*)d_in[13];
    const float* fcb    = (const float*)d_in[14];
    float* out = (float*)d_out;

    char* ws = (char*)d_ws;
    __bf16*   Bt    = (__bf16*)(ws + O_BT);
    __bf16*   rootT = (__bf16*)(ws + O_ROOT);
    __bf16*   gatT  = (__bf16*)(ws + O_GAT);
    float*    agg   = (float*)(ws + O_AGG);
    float*    x1    = (float*)(ws + O_X1);
    float*    xp    = (float*)(ws + O_XP);
    float*    outA  = (float*)(ws + O_OUT);
    float*    as_   = (float*)(ws + O_AS);
    float*    an_   = (float*)(ws + O_AN);
    unsigned* smK   = (unsigned*)(ws + O_SMK);
    float*    den   = (float*)(ws + O_DEN);
    float*    sc    = (float*)(ws + O_SC);
    float*    ex    = (float*)(ws + O_EX);
    float*    pool  = (float*)(ws + O_POOL);

    k_prep <<<(64 * KEXT + 255) / 256, 256, 0, stream>>>(w1, b1, root, gk, Bt, rootT, gatT);
    k_init <<<(NN * 64 + 255) / 256, 256, 0, stream>>>(agg, outA, den, smK, pool);
    k_ecc  <<<(NE + 127) / 128, 256, ECC_SMEM, stream>>>(x, e, eidx, w0, b0, Bt, agg);
    k_node1<<<(NN / 16 + 7) / 8, 256, 0, stream>>>(x, agg, rootT, ebias, x1);
    k_node2<<<(NN / 16 + 7) / 8, 256, 0, stream>>>(x1, gatT, xp);
    k_attn <<<(NN + 255) / 256, 256, 0, stream>>>(xp, wself, wneigh, as_, an_);
    k_emax <<<(NE + 255) / 256, 256, 0, stream>>>(eidx, as_, an_, sc, smK);
    k_eexp <<<(NE + 255) / 256, 256, 0, stream>>>(eidx, sc, smK, ex, den);
    k_escat<<<(int)(((long long)NE * 64 + 255) / 256), 256, 0, stream>>>(eidx, ex, den, xp, outA);
    k_pool <<<(NN + 255) / 256, 64, 0, stream>>>(outA, gbias, pool);
    k_dense<<<1, 32, 0, stream>>>(pool, fcw, fcb, out);
}
